// selfAttention_83743272337759
// MI455X (gfx1250) — matrix operand order repaired and checked by reading, not compiled, not run on hardware
//
#include <hip/hip_runtime.h>
#include <hip/hip_bf16.h>

// ---------------------------------------------------------------------------
// Self-attention (B=4, N=2048, D=1024, no head split) for gfx1250 / MI455X.
// Pipeline:
//   1) cvt   : x, W_qkv  f32 -> bf16
//   2) qkv   : Y = X @ W^T  (WMMA bf16).  Epilogue scatters to Q(*scale), K,
//              and V^T (transposed store so P@V is K-contiguous too).
//   3) per batch: S = Q @ K^T (f32 out) -> row softmax -> P (bf16)
//                 out = P @ V   (WMMA bf16, f32 out)
// All matrix math via v_wmma_f32_16x16x32_bf16.
// Per-wave tile: 32(M) x 64(N) = 2 A-frags x 4 B-frags -> 8 accumulators,
// 21.3 FLOP per loaded byte (vs 12.8 for 16x64).
// ---------------------------------------------------------------------------

typedef __attribute__((ext_vector_type(4)))  __bf16 v4bf;
typedef __attribute__((ext_vector_type(8)))  __bf16 v8bf;
typedef __attribute__((ext_vector_type(16))) __bf16 v16bf;
typedef __attribute__((ext_vector_type(8)))  float  v8f;

static constexpr int BATCH = 4;
static constexpr int SEQ   = 2048;
static constexpr int DIM   = 1024;
static constexpr int EDIM  = 3 * DIM;

// Workspace layout (bytes). Total ~94 MiB.
static constexpr size_t OFF_WB = 0;                                       // bf16 [3072][1024]
static constexpr size_t OFF_XB = OFF_WB + (size_t)EDIM * DIM * 2;         // bf16 [4][2048][1024]
static constexpr size_t OFF_QB = OFF_XB + (size_t)BATCH * SEQ * DIM * 2;  // bf16 [4][2048][1024] (pre-scaled)
static constexpr size_t OFF_KB = OFF_QB + (size_t)BATCH * SEQ * DIM * 2;  // bf16 [4][2048][1024]
static constexpr size_t OFF_VT = OFF_KB + (size_t)BATCH * SEQ * DIM * 2;  // bf16 [4][1024][2048] (V transposed)
static constexpr size_t OFF_S  = OFF_VT + (size_t)BATCH * SEQ * DIM * 2;  // f32  [2048][2048] (one batch)
static constexpr size_t OFF_P  = OFF_S  + (size_t)SEQ * SEQ * 4;          // bf16 [2048][2048] (one batch)
static constexpr size_t WS_NEED = OFF_P + (size_t)SEQ * SEQ * 2;

// --- fragment loaders (wave32, contiguous k = 16h + s per lane half) -------------------------

// A: 16x32 bf16 (MxK), row-major source, K contiguous.
__device__ __forceinline__ v16bf load_a_frag(const __bf16* A, int lda,
                                             int mBase, int k0, int lane) {
  const int row = mBase + (lane & 15);
  const int kb  = k0 + ((lane >> 4) << 4);
  const __bf16* p = A + (size_t)row * lda + kb;
  v8bf lo = *(const v8bf*)(p);
  v8bf hi = *(const v8bf*)(p + 8);
  return __builtin_shufflevector(lo, hi, 0,1,2,3,4,5,6,7,8,9,10,11,12,13,14,15);
}

// B: 32x16 bf16 (KxN) where B[k][n] = Bm[n][k] (Bm row-major, K contiguous).
__device__ __forceinline__ v16bf load_b_frag(const __bf16* Bm, int ldb,
                                             int nBase, int k0, int lane) {
  const int col = nBase + (lane & 15);
  const int kb  = k0 + ((lane >> 4) << 4);
  return *(const v16bf*)(Bm + (size_t)col * ldb + kb);   // 32B, 32B-aligned
}

// Generic 32x64-per-wave WMMA inner product over K, A/B both K-contiguous.
// acc[mi][t] covers rows mBase+mi*16.. and cols nBase+t*16..
template <int KLEN>
__device__ __forceinline__ void gemm_tile_32x64(const __bf16* __restrict__ A, int lda,
                                                const __bf16* __restrict__ B, int ldb,
                                                int mBase, int nBase, int lane,
                                                v8f acc[2][4]) {
  #pragma unroll 2
  for (int k0 = 0; k0 < KLEN; k0 += 32) {
    v16bf af0 = load_a_frag(A, lda, mBase,      k0, lane);
    v16bf af1 = load_a_frag(A, lda, mBase + 16, k0, lane);
    #pragma unroll
    for (int t = 0; t < 4; ++t) {
      v16bf bf = load_b_frag(B, ldb, nBase + t * 16, k0, lane);
      acc[0][t] = __builtin_amdgcn_wmma_f32_16x16x32_bf16(false, af0, false, bf,
                                                          (short)0, acc[0][t], false, false);
      acc[1][t] = __builtin_amdgcn_wmma_f32_16x16x32_bf16(false, af1, false, bf,
                                                          (short)0, acc[1][t], false, false);
    }
  }
}

// --- kernels ---------------------------------------------------------------

__global__ __launch_bounds__(256) void cvt_f32_bf16(const float4* __restrict__ in,
                                                    v4bf* __restrict__ out, int n4) {
  int i = blockIdx.x * 256 + threadIdx.x;
  if (i < n4) {
    float4 f = in[i];
    v4bf o = { (__bf16)f.x, (__bf16)f.y, (__bf16)f.z, (__bf16)f.w };
    out[i] = o;
  }
}

// Y = X[b] @ W^T ; scatter epilogue into Q (scaled), K, V^T.
// Workgroup tile 64x256 (8 waves, 2x4).  Grid: (12, 32, 4), 256 thr.
__global__ __launch_bounds__(256) void qkv_gemm(const __bf16* __restrict__ Xb,
                                                const __bf16* __restrict__ Wb,
                                                __bf16* __restrict__ Qb,
                                                __bf16* __restrict__ Kb,
                                                __bf16* __restrict__ Vt) {
  const int lane  = threadIdx.x & 31;
  const int wave  = threadIdx.x >> 5;                 // 0..7  (2 x 4 waves)
  const int b     = blockIdx.z;
  const int mBase = blockIdx.y * 64 + (wave >> 2) * 32;
  const int nBase = blockIdx.x * 256 + (wave & 3) * 64;
  const __bf16* A = Xb + (size_t)b * SEQ * DIM;

  v8f acc[2][4] = {};
  gemm_tile_32x64<DIM>(A, DIM, Wb, DIM, mBase, nBase, lane, acc);

  const float scale = 0.03125f;                       // DIM^-0.5
  const int rOff = (lane >> 4) << 3;
  #pragma unroll
  for (int mi = 0; mi < 2; ++mi) {
    #pragma unroll
    for (int t = 0; t < 4; ++t) {
      #pragma unroll
      for (int j = 0; j < 8; ++j) {
        const int row = mBase + mi * 16 + j + rOff;
        const int e   = nBase + t * 16 + (lane & 15);
        const float y = acc[mi][t][j];
        if (e < DIM) {
          Qb[((size_t)b * SEQ + row) * DIM + e] = (__bf16)(y * scale);
        } else if (e < 2 * DIM) {
          Kb[((size_t)b * SEQ + row) * DIM + (e - DIM)] = (__bf16)y;
        } else {
          Vt[((size_t)b * DIM + (e - 2 * DIM)) * SEQ + row] = (__bf16)y;
        }
      }
    }
  }
}

// S = Qs @ K^T ; f32 out.  Grid: (8, 32), 256 thr.
__global__ __launch_bounds__(256) void qk_gemm(const __bf16* __restrict__ Q,
                                               const __bf16* __restrict__ K,
                                               float* __restrict__ S) {
  const int lane  = threadIdx.x & 31;
  const int wave  = threadIdx.x >> 5;
  const int mBase = blockIdx.y * 64 + (wave >> 2) * 32;
  const int nBase = blockIdx.x * 256 + (wave & 3) * 64;

  v8f acc[2][4] = {};
  gemm_tile_32x64<DIM>(Q, DIM, K, DIM, mBase, nBase, lane, acc);

  const int rOff = (lane >> 4) << 3;
  #pragma unroll
  for (int mi = 0; mi < 2; ++mi)
    #pragma unroll
    for (int t = 0; t < 4; ++t)
      #pragma unroll
      for (int j = 0; j < 8; ++j)
        S[(size_t)(mBase + mi * 16 + j + rOff) * SEQ + nBase + t * 16 + (lane & 15)]
            = acc[mi][t][j];
}

// Row softmax: one 256-thread block per row of S[2048]; writes bf16 P.
__global__ __launch_bounds__(256) void softmax_rows(const float* __restrict__ S,
                                                    __bf16* __restrict__ P) {
  __shared__ float red[256];
  const int row = blockIdx.x;
  const int tid = threadIdx.x;
  const float* s = S + (size_t)row * SEQ;

  float vals[8];
  float m = -3.4e38f;
  #pragma unroll
  for (int i = 0; i < 8; ++i) { vals[i] = s[tid + i * 256]; m = fmaxf(m, vals[i]); }
  red[tid] = m; __syncthreads();
  for (int off = 128; off > 0; off >>= 1) {
    if (tid < off) red[tid] = fmaxf(red[tid], red[tid + off]);
    __syncthreads();
  }
  m = red[0]; __syncthreads();

  float sum = 0.f;
  #pragma unroll
  for (int i = 0; i < 8; ++i) { vals[i] = __expf(vals[i] - m); sum += vals[i]; }
  red[tid] = sum; __syncthreads();
  for (int off = 128; off > 0; off >>= 1) {
    if (tid < off) red[tid] += red[tid + off];
    __syncthreads();
  }
  const float inv = 1.0f / red[0];
  #pragma unroll
  for (int i = 0; i < 8; ++i)
    P[(size_t)row * SEQ + tid + i * 256] = (__bf16)(vals[i] * inv);
}

// out = P @ V  with V stored transposed: out[n][d] = sum_m P[n][m] * Vt[d][m].
// Grid: (4, 32), 256 thr.
__global__ __launch_bounds__(256) void av_gemm(const __bf16* __restrict__ P,
                                               const __bf16* __restrict__ Vt,
                                               float* __restrict__ Out) {
  const int lane  = threadIdx.x & 31;
  const int wave  = threadIdx.x >> 5;
  const int mBase = blockIdx.y * 64 + (wave >> 2) * 32;
  const int nBase = blockIdx.x * 256 + (wave & 3) * 64;   // over DIM

  v8f acc[2][4] = {};
  gemm_tile_32x64<SEQ>(P, SEQ, Vt, SEQ, mBase, nBase, lane, acc);

  const int rOff = (lane >> 4) << 3;
  #pragma unroll
  for (int mi = 0; mi < 2; ++mi)
    #pragma unroll
    for (int t = 0; t < 4; ++t)
      #pragma unroll
      for (int j = 0; j < 8; ++j)
        Out[(size_t)(mBase + mi * 16 + j + rOff) * DIM + nBase + t * 16 + (lane & 15)]
            = acc[mi][t][j];
}

// ---------------------------------------------------------------------------

extern "C" void kernel_launch(void* const* d_in, const int* in_sizes, int n_in,
                              void* d_out, int out_size, void* d_ws, size_t ws_size,
                              hipStream_t stream) {
  if (ws_size < WS_NEED) return;

  const float* x = (const float*)d_in[0];   // [4][2048][1024] f32
  const float* w = (const float*)d_in[1];   // [3072][1024]    f32
  float* out = (float*)d_out;               // [4][2048][1024] f32
  char* ws = (char*)d_ws;

  __bf16* Wb = (__bf16*)(ws + OFF_WB);
  __bf16* Xb = (__bf16*)(ws + OFF_XB);
  __bf16* Qb = (__bf16*)(ws + OFF_QB);
  __bf16* Kb = (__bf16*)(ws + OFF_KB);
  __bf16* Vt = (__bf16*)(ws + OFF_VT);
  float*  S  = (float*)(ws + OFF_S);
  __bf16* P  = (__bf16*)(ws + OFF_P);

  // 1) downconvert inputs to bf16
  {
    int n4 = BATCH * SEQ * DIM / 4;
    cvt_f32_bf16<<<dim3((n4 + 255) / 256), dim3(256), 0, stream>>>(
        (const float4*)x, (v4bf*)Xb, n4);
    int w4 = EDIM * DIM / 4;
    cvt_f32_bf16<<<dim3((w4 + 255) / 256), dim3(256), 0, stream>>>(
        (const float4*)w, (v4bf*)Wb, w4);
  }

  // 2) fused QKV projection (all batches)
  qkv_gemm<<<dim3(EDIM / 256, SEQ / 64, BATCH), dim3(256), 0, stream>>>(
      Xb, Wb, Qb, Kb, Vt);

  // 3) per-batch attention (S and P buffers reused; stream order serializes)
  for (int b = 0; b < BATCH; ++b) {
    const __bf16* Qs = Qb + (size_t)b * SEQ * DIM;
    const __bf16* Ks = Kb + (size_t)b * SEQ * DIM;
    const __bf16* Vs = Vt + (size_t)b * DIM * SEQ;
    float* Os = out + (size_t)b * SEQ * DIM;

    qk_gemm<<<dim3(SEQ / 256, SEQ / 64), dim3(256), 0, stream>>>(Qs, Ks, S);
    softmax_rows<<<dim3(SEQ), dim3(256), 0, stream>>>(S, P);
    av_gemm<<<dim3(DIM / 256, SEQ / 64), dim3(256), 0, stream>>>(P, Vs, Os);
  }
}
